// Encoder_A_25967372271948
// MI455X (gfx1250) — matrix operand order repaired and checked by reading, not compiled, not run on hardware
//
#include <hip/hip_runtime.h>
#include <hip/hip_bf16.h>
#include <math.h>

typedef __attribute__((ext_vector_type(16))) _Float16 v16h;
typedef __attribute__((ext_vector_type(8)))  float    v8f;

#define BM 128
#define BN 128
#define BK 32
#define LDSTR 28   // dwords per LDS tile row: 16 data dwords + 12 pad (112B, 16B-aligned)

// ---------------------------------------------------------------------------
// gfx1250 async global->LDS staging (ASYNCcnt path, ISA 15.18.3 / 08_async_tensor)
// ---------------------------------------------------------------------------

static __device__ __forceinline__ void async_ld_b128(uint32_t lds_byte, const void* gptr) {
  asm volatile("global_load_async_to_lds_b128 %0, %1, off"
               :: "v"(lds_byte), "v"((uint64_t)(uintptr_t)gptr)
               : "memory");
}
static __device__ __forceinline__ void wait_async0() {
  asm volatile("s_wait_asynccnt 0" ::: "memory");
}

// ---------------------------------------------------------------------------
// elementwise / layout kernels
// ---------------------------------------------------------------------------

__global__ __launch_bounds__(256) void k_cast_f2h(const float* __restrict__ s,
                                                  _Float16* __restrict__ d, int n) {
  int i = blockIdx.x * blockDim.x + threadIdx.x;
  if (i < n) d[i] = (_Float16)s[i];
}

// activation f32 [M,K] -> f16 [M,Kpad], zero-filled K tail
__global__ __launch_bounds__(256) void k_cast_pad(const float* __restrict__ s,
                                                  _Float16* __restrict__ d,
                                                  int M, int K, int Kpad) {
  int i = blockIdx.x * blockDim.x + threadIdx.x;
  if (i >= M * Kpad) return;
  int m = i / Kpad, k = i - m * Kpad;
  d[i] = (_Float16)((k < K) ? s[(size_t)m * K + k] : 0.f);
}

// weight f32 [N,K] -> f16 [Npad,Kpad], zero pad
__global__ __launch_bounds__(256) void k_castw_pad(const float* __restrict__ s,
                                                   _Float16* __restrict__ d,
                                                   int Nn, int K, int Npad, int Kpad) {
  int i = blockIdx.x * blockDim.x + threadIdx.x;
  if (i >= Npad * Kpad) return;
  int n = i / Kpad, k = i - n * Kpad;
  d[i] = (_Float16)((n < Nn && k < K) ? s[(size_t)n * K + k] : 0.f);
}

// dw_w [hidden,hidden,3,3] (o,c,ky,kx) -> f16 [Npad][tap][Kpad], tap = ky*3+kx, zero pad
__global__ __launch_bounds__(256) void k_packdw_pad(const float* __restrict__ s,
                                                    _Float16* __restrict__ d,
                                                    int hidden, int Npad, int Kpad) {
  int i = blockIdx.x * blockDim.x + threadIdx.x;
  if (i >= Npad * 9 * Kpad) return;
  int o = i / (9 * Kpad);
  int r = i - o * (9 * Kpad);
  int t = r / Kpad;
  int c = r - t * Kpad;
  d[i] = (_Float16)((o < hidden && c < hidden) ? s[((size_t)o * hidden + c) * 9 + t] : 0.f);
}

__global__ __launch_bounds__(256) void k_nhwc_to_nchw(const float* __restrict__ in,
                                                      float* __restrict__ out, int C, int HW) {
  int i = blockIdx.x * blockDim.x + threadIdx.x;
  if (i >= C * HW) return;
  int c = i / HW, p = i - c * HW;
  out[i] = in[(size_t)p * C + c];
}

// ---------------------------------------------------------------------------
// direct 3x3 convs (cheap, memory-bound stages)
// ---------------------------------------------------------------------------

__global__ __launch_bounds__(256) void k_conv3x3_plain(const float* __restrict__ in,
                                                       const float* __restrict__ w,
                                                       float* __restrict__ out,
                                                       int Cin, int Cout, int H, int W) {
  int i = blockIdx.x * blockDim.x + threadIdx.x;
  int HW = H * W;
  if (i >= Cout * HW) return;
  int co = i / HW; int p = i - co * HW; int y = p / W, x = p - y * W;
  float acc = 0.f;
  for (int ci = 0; ci < Cin; ++ci) {
    const float* ip = in + (size_t)ci * HW;
    const float* wp = w + ((size_t)co * Cin + ci) * 9;
#pragma unroll
    for (int ky = 0; ky < 3; ++ky) {
      int sy = y + ky - 1; if ((unsigned)sy >= (unsigned)H) continue;
#pragma unroll
      for (int kx = 0; kx < 3; ++kx) {
        int sx = x + kx - 1; if ((unsigned)sx >= (unsigned)W) continue;
        acc += ip[sy * W + sx] * wp[ky * 3 + kx];
      }
    }
  }
  out[i] = acc;
}

// 3x3 conv (pad 1) fused with pixel_unshuffle(2). out channel = c*4 + (y&1)*2 + (x&1)
__global__ __launch_bounds__(256) void k_conv3x3_down(const float* __restrict__ in,
                                                      const float* __restrict__ w,
                                                      float* __restrict__ out,
                                                      int Cin, int Cout, int H, int W,
                                                      int in_nhwc, int out_nhwc) {
  int i = blockIdx.x * blockDim.x + threadIdx.x;
  int HW = H * W;
  if (i >= Cout * HW) return;
  int c = i / HW; int p = i - c * HW; int y = p / W, x = p - y * W;
  float acc = 0.f;
  for (int ci = 0; ci < Cin; ++ci) {
    const float* wp = w + ((size_t)c * Cin + ci) * 9;
#pragma unroll
    for (int ky = 0; ky < 3; ++ky) {
      int sy = y + ky - 1; if ((unsigned)sy >= (unsigned)H) continue;
#pragma unroll
      for (int kx = 0; kx < 3; ++kx) {
        int sx = x + kx - 1; if ((unsigned)sx >= (unsigned)W) continue;
        float v = in_nhwc ? in[(size_t)(sy * W + sx) * Cin + ci]
                          : in[(size_t)ci * HW + sy * W + sx];
        acc += v * wp[ky * 3 + kx];
      }
    }
  }
  int oc = c * 4 + ((y & 1) << 1) + (x & 1);
  int Wo = W >> 1, Ho = H >> 1, oy = y >> 1, ox = x >> 1;
  if (out_nhwc) out[(size_t)(oy * Wo + ox) * (Cout * 4) + oc] = acc;
  else          out[(size_t)oc * (Ho * Wo) + oy * Wo + ox]    = acc;
}

// ---------------------------------------------------------------------------
// per-pixel channel LayerNorm (NHWC) -> f16 output (GEMM A operand)
// ---------------------------------------------------------------------------

__global__ __launch_bounds__(256) void k_layernorm(const float* __restrict__ X,
                                                   const float* __restrict__ g,
                                                   const float* __restrict__ b,
                                                   _Float16* __restrict__ Y, int M, int C) {
  int wv = blockIdx.x * (blockDim.x >> 5) + (threadIdx.x >> 5);
  if (wv >= M) return;
  int lane = threadIdx.x & 31;
  const float* x = X + (size_t)wv * C;
  float s = 0.f, s2 = 0.f;
  for (int c = lane; c < C; c += 32) { float v = x[c]; s += v; s2 += v * v; }
#pragma unroll
  for (int o = 16; o > 0; o >>= 1) { s += __shfl_xor(s, o, 32); s2 += __shfl_xor(s2, o, 32); }
  float mu  = s / (float)C;
  float var = s2 / (float)C - mu * mu;
  float inv = rsqrtf(var + 1e-5f);
  _Float16* y = Y + (size_t)wv * C;
  for (int c = lane; c < C; c += 32)
    y[c] = (_Float16)(((x[c] - mu) * inv) * g[c] + b[c]);
}

// ---------------------------------------------------------------------------
// multi-dilate windowed attention: wave per (pixel, dilation-group, head)
// lane = head-dim channel (head_dim == 32 == wave32)
// ---------------------------------------------------------------------------

__global__ __launch_bounds__(256) void k_attn(const float* __restrict__ QKV,
                                              float* __restrict__ Out, int H, int W, int C) {
  int M = H * W;
  int d = C >> 1;
  int hpg = d >> 5;
  int units = 2 * hpg;
  int wv = blockIdx.x * (blockDim.x >> 5) + (threadIdx.x >> 5);
  if (wv >= M * units) return;
  int lane = threadIdx.x & 31;
  int pix = wv / units;
  int u   = wv - pix * units;
  int grp = u / hpg, h = u - grp * hpg;
  int dil = grp ? 3 : 1;
  int y0 = pix / W, x0 = pix - y0 * W;
  int cbase = grp * d + h * 32 + lane;
  int C3 = 3 * C;
  float q = QKV[(size_t)pix * C3 + cbase];
  const float scale = 0.1767766952966369f;  // 32^-0.5
  float a[9];
#pragma unroll
  for (int t = 0; t < 9; ++t) {
    int sy = y0 + (t / 3 - 1) * dil, sx = x0 + (t % 3 - 1) * dil;
    float kv = 0.f;
    if ((unsigned)sy < (unsigned)H && (unsigned)sx < (unsigned)W)
      kv = QKV[(size_t)(sy * W + sx) * C3 + C + cbase];
    float p = q * kv;
#pragma unroll
    for (int o = 16; o > 0; o >>= 1) p += __shfl_xor(p, o, 32);
    a[t] = p * scale;
  }
  float mx = a[0];
#pragma unroll
  for (int t = 1; t < 9; ++t) mx = fmaxf(mx, a[t]);
  float se = 0.f;
#pragma unroll
  for (int t = 0; t < 9; ++t) { a[t] = __expf(a[t] - mx); se += a[t]; }
  float inv = 1.f / se;
  float o = 0.f;
#pragma unroll
  for (int t = 0; t < 9; ++t) {
    int sy = y0 + (t / 3 - 1) * dil, sx = x0 + (t % 3 - 1) * dil;
    float vv = 0.f;
    if ((unsigned)sy < (unsigned)H && (unsigned)sx < (unsigned)W)
      vv = QKV[(size_t)(sy * W + sx) * C3 + 2 * C + cbase];
    o += a[t] * inv * vv;
  }
  Out[(size_t)pix * C + cbase] = o;
}

// ---------------------------------------------------------------------------
// WMMA implicit GEMM, 128x128 block tile, 8 waves x (32x64) wave tiles.
// A: f16 NHWC [M,Kpad] (zero-padded K tail). Bw: f16 [Npad][ntaps*Kpad] (zero-padded).
// ntaps==1: plain GEMM, both tiles staged with global_load_async_to_lds_b128.
// ntaps==9: 3x3 conv (pad 1); A staged branchlessly (clamp + select), B async.
// Epilogue fuses bias, exact GELU, residual add.
// ---------------------------------------------------------------------------

static __device__ __forceinline__ v16h frag_a(const uint32_t* s, int row0, int lane) {
  // ISA 7.12.2: A lane = M (lane&15); K halves {0-7,16-23} / {8-15,24-31}
  union { v16h h; uint32_t u[8]; } r;
  const uint32_t* p = s + (row0 + (lane & 15)) * LDSTR + ((lane >> 4) << 2);
#pragma unroll
  for (int v = 0; v < 8; ++v) r.u[v] = p[(v & 3) + ((v >> 2) << 3)];
  return r.h;
}

static __device__ __forceinline__ v16h frag_b(const uint32_t* s, int row0, int lane) {
  // B lane = N (lane&15); K halves {0-7,16-23} / {8-15,24-31}
  union { v16h h; uint32_t u[8]; } r;
  const uint32_t* p = s + (row0 + (lane & 15)) * LDSTR + ((lane >> 4) << 2);
#pragma unroll
  for (int v = 0; v < 8; ++v) r.u[v] = p[(v & 3) + ((v >> 2) << 3)];
  return r.h;
}

__global__ __launch_bounds__(256) void k_gemm_wmma(const _Float16* __restrict__ A,
                                                   const _Float16* __restrict__ Bw,
                                                   const float* __restrict__ bias,
                                                   const float* __restrict__ resid,
                                                   float* __restrict__ Out,
                                                   int H, int W, int Kpad, int N,
                                                   int ntaps, int fuse_gelu) {
  __shared__ __align__(16) uint32_t sA[BM * LDSTR];
  __shared__ __align__(16) uint32_t sB[BN * LDSTR];
  const int M = H * W;
  const int tid = threadIdx.x;
  const int lane = tid & 31, wv = tid >> 5;
  const int mBlock = blockIdx.y * BM;
  const int nBlock = blockIdx.x * BN;
  const int wm = (wv >> 1) << 5;   // 0,32,64,96
  const int wn = (wv & 1) << 6;    // 0,64
  const int ldb = ntaps * Kpad;
  const int kSteps = Kpad / BK;
  const uint32_t sAoff = (uint32_t)(uintptr_t)(&sA[0]);
  const uint32_t sBoff = (uint32_t)(uintptr_t)(&sB[0]);

  v8f zero8 = {0.f, 0.f, 0.f, 0.f, 0.f, 0.f, 0.f, 0.f};
  v8f acc[2][4] = {{zero8, zero8, zero8, zero8}, {zero8, zero8, zero8, zero8}};

  for (int tap = 0; tap < ntaps; ++tap) {
    const int dy = (ntaps == 9) ? (tap / 3 - 1) : 0;
    const int dx = (ntaps == 9) ? (tap % 3 - 1) : 0;
    for (int kb = 0; kb < kSteps; ++kb) {
      const int k0 = kb * BK;
      // ---- B tile [BN x BK]: async copy, guard-free (padded weights) ----
#pragma unroll
      for (int i = 0; i < 2; ++i) {
        int idx = i * 256 + tid;
        int row = idx >> 2, q = idx & 3;
        uint32_t lo = sBoff + (uint32_t)(row * LDSTR + (q << 2)) * 4u;
        async_ld_b128(lo, (const void*)(Bw + (size_t)(nBlock + row) * ldb + tap * Kpad + k0 + (q << 3)));
      }
      if (ntaps == 1) {
        // ---- A tile: async copy, guard-free (M multiple of 128, padded K) ----
#pragma unroll
        for (int i = 0; i < 2; ++i) {
          int idx = i * 256 + tid;
          int row = idx >> 2, q = idx & 3;
          uint32_t lo = sAoff + (uint32_t)(row * LDSTR + (q << 2)) * 4u;
          async_ld_b128(lo, (const void*)(A + (size_t)(mBlock + row) * Kpad + k0 + (q << 3)));
        }
      } else {
        // ---- A tile with spatial shift: branchless clamp + select ----
#pragma unroll
        for (int i = 0; i < 8; ++i) {
          int idx = i * 256 + tid;
          int row = idx >> 4, dcol = idx & 15;
          int kk = k0 + (dcol << 1);
          int m = mBlock + row;
          int y = m / W, x = m - y * W;
          int sy = y + dy, sx = x + dx;
          bool ok = ((unsigned)sy < (unsigned)H) && ((unsigned)sx < (unsigned)W);
          int cy = sy < 0 ? 0 : (sy >= H ? H - 1 : sy);
          int cx = sx < 0 ? 0 : (sx >= W ? W - 1 : sx);
          uint32_t v = *(const uint32_t*)(A + (size_t)(cy * W + cx) * Kpad + kk);
          sA[row * LDSTR + dcol] = ok ? v : 0u;
        }
      }
      wait_async0();
      __syncthreads();

      v16h a0 = frag_a(sA, wm,      lane);
      v16h a1 = frag_a(sA, wm + 16, lane);
      v16h b0 = frag_b(sB, wn,      lane);
      v16h b1 = frag_b(sB, wn + 16, lane);
      v16h b2 = frag_b(sB, wn + 32, lane);
      v16h b3 = frag_b(sB, wn + 48, lane);

      acc[0][0] = __builtin_amdgcn_wmma_f32_16x16x32_f16(false, a0, false, b0, (short)0, acc[0][0], false, false);
      acc[0][1] = __builtin_amdgcn_wmma_f32_16x16x32_f16(false, a0, false, b1, (short)0, acc[0][1], false, false);
      acc[0][2] = __builtin_amdgcn_wmma_f32_16x16x32_f16(false, a0, false, b2, (short)0, acc[0][2], false, false);
      acc[0][3] = __builtin_amdgcn_wmma_f32_16x16x32_f16(false, a0, false, b3, (short)0, acc[0][3], false, false);
      acc[1][0] = __builtin_amdgcn_wmma_f32_16x16x32_f16(false, a1, false, b0, (short)0, acc[1][0], false, false);
      acc[1][1] = __builtin_amdgcn_wmma_f32_16x16x32_f16(false, a1, false, b1, (short)0, acc[1][1], false, false);
      acc[1][2] = __builtin_amdgcn_wmma_f32_16x16x32_f16(false, a1, false, b2, (short)0, acc[1][2], false, false);
      acc[1][3] = __builtin_amdgcn_wmma_f32_16x16x32_f16(false, a1, false, b3, (short)0, acc[1][3], false, false);
      __syncthreads();
    }
  }

  // ---- epilogue: C/D layout -> global, fuse bias / GELU / residual ----
#pragma unroll
  for (int mi = 0; mi < 2; ++mi) {
#pragma unroll
    for (int ni = 0; ni < 4; ++ni) {
      union { v8f v; float f[8]; } t;
      t.v = acc[mi][ni];
#pragma unroll
      for (int r = 0; r < 8; ++r) {
        int m = mBlock + wm + mi * 16 + ((lane >> 4) << 3) + r;
        int n = nBlock + wn + ni * 16 + (lane & 15);
        if (m < M && n < N) {
          float val = t.f[r];
          if (bias) val += bias[n];
          if (fuse_gelu) val = 0.5f * val * (1.f + erff(val * 0.70710678118654752f));
          if (resid) val += resid[(size_t)m * N + n];
          Out[(size_t)m * N + n] = val;
        }
      }
    }
  }
}

// ---------------------------------------------------------------------------
// host-side orchestration
// ---------------------------------------------------------------------------

static inline int c256(int n) { return (n + 255) / 256; }

static void run_block(void* const* P, float* act, int H, int W, int C, int hidden,
                      _Float16* a16, _Float16* h16, float* S1, float* S2,
                      _Float16* wq, _Float16* wpr, _Float16* wpin, _Float16* wdw,
                      _Float16* wpo, hipStream_t st) {
  const float* n1w = (const float*)P[0];
  const float* n1b = (const float*)P[1];
  const float* qkv = (const float*)P[2];
  const float* prw = (const float*)P[3];
  const float* prb = (const float*)P[4];
  const float* n2w = (const float*)P[5];
  const float* n2b = (const float*)P[6];
  const float* pin = (const float*)P[7];
  const float* dww = (const float*)P[8];
  const float* pou = (const float*)P[9];
  const int M = H * W;
  const int Hpad  = (hidden + 31) & ~31;    // K padding (mult of BK)
  const int HNpad = (hidden + 127) & ~127;  // N padding (mult of BN)
  const dim3 blk(256);

  // per-block weight conversion / repack (zero-padded to tile multiples)
  k_castw_pad<<<c256(3 * C * C), blk, 0, st>>>(qkv, wq, 3 * C, C, 3 * C, C);
  k_castw_pad<<<c256(C * C), blk, 0, st>>>(prw, wpr, C, C, C, C);
  k_castw_pad<<<c256(HNpad * C), blk, 0, st>>>(pin, wpin, hidden, C, HNpad, C);
  k_packdw_pad<<<c256(HNpad * 9 * Hpad), blk, 0, st>>>(dww, wdw, hidden, HNpad, Hpad);
  k_castw_pad<<<c256(C * Hpad), blk, 0, st>>>(pou, wpo, C, hidden, C, Hpad);

  // ---- attention branch: x += proj(attn(qkv(LN1(x)))) ----
  k_layernorm<<<dim3((M + 7) / 8), blk, 0, st>>>(act, n1w, n1b, a16, M, C);
  k_gemm_wmma<<<dim3((3 * C + BN - 1) / BN, M / BM), blk, 0, st>>>(
      a16, wq, nullptr, nullptr, S1, H, W, C, 3 * C, 1, 0);
  {
    int units = C / 32;  // 2 groups * (C/2/32) heads
    k_attn<<<dim3((M * units + 7) / 8), blk, 0, st>>>(S1, S2, H, W, C);
  }
  k_cast_f2h<<<c256(M * C), blk, 0, st>>>(S2, a16, M * C);
  k_gemm_wmma<<<dim3((C + BN - 1) / BN, M / BM), blk, 0, st>>>(
      a16, wpr, prb, act, act, H, W, C, C, 1, 0);

  // ---- FFN branch: x += pout(gelu(dw3x3(pin(LN2(x))))) ----
  k_layernorm<<<dim3((M + 7) / 8), blk, 0, st>>>(act, n2w, n2b, a16, M, C);
  k_gemm_wmma<<<dim3((hidden + BN - 1) / BN, M / BM), blk, 0, st>>>(
      a16, wpin, nullptr, nullptr, S1, H, W, C, hidden, 1, 0);
  k_cast_pad<<<c256(M * Hpad), blk, 0, st>>>(S1, h16, M, hidden, Hpad);
  k_gemm_wmma<<<dim3((hidden + BN - 1) / BN, M / BM), blk, 0, st>>>(
      h16, wdw, nullptr, nullptr, S2, H, W, Hpad, hidden, 9, 1);
  k_cast_pad<<<c256(M * Hpad), blk, 0, st>>>(S2, h16, M, hidden, Hpad);
  k_gemm_wmma<<<dim3((C + BN - 1) / BN, M / BM), blk, 0, st>>>(
      h16, wpo, nullptr, act, act, H, W, Hpad, C, 1, 0);
}

extern "C" void kernel_launch(void* const* d_in, const int* in_sizes, int n_in,
                              void* d_out, int out_size, void* d_ws, size_t ws_size,
                              hipStream_t stream) {
  (void)in_sizes; (void)n_in; (void)out_size; (void)ws_size;
  const float* inp   = (const float*)d_in[0];
  const float* pe_w  = (const float*)d_in[1];
  const float* d12_w = (const float*)d_in[2];
  const float* d23_w = (const float*)d_in[3];
  const float* d34_w = (const float*)d_in[4];

  float* out = (float*)d_out;
  float* x4o = out;                        // [256,48,48]
  float* x3o = x4o + 256 * 48 * 48;        // [128,96,96]
  float* x2o = x3o + 128 * 96 * 96;        // [64,192,192]
  float* x1o = x2o + 64 * 192 * 192;       // [32,384,384]

  // workspace arena (deterministic layout)
  char* base = (char*)d_ws;
  size_t off = 0;
  auto arena = [&](size_t bytes) -> char* {
    char* p = base + off;
    off += (bytes + 255) & ~(size_t)255;
    return p;
  };
  float*    act3 = (float*)arena((size_t)9216 * 128 * 4);
  float*    act4 = (float*)arena((size_t)2304 * 256 * 4);
  _Float16* a16  = (_Float16*)arena((size_t)9216 * 256 * 2);
  _Float16* h16  = (_Float16*)arena((size_t)9216 * 352 * 2);   // [M, Hpad] max
  float*    S1   = (float*)arena((size_t)9216 * 384 * 4);
  float*    S2   = (float*)arena((size_t)9216 * 340 * 4);
  _Float16* wq   = (_Float16*)arena((size_t)768 * 256 * 2);    // [3C][C] max
  _Float16* wpr  = (_Float16*)arena((size_t)256 * 256 * 2);    // [C][C] max
  _Float16* wpin = (_Float16*)arena((size_t)768 * 256 * 2);    // [HNpad][C] max
  _Float16* wdw  = (_Float16*)arena((size_t)768 * 9 * 704 * 2);// [HNpad][9*Hpad] max
  _Float16* wpo  = (_Float16*)arena((size_t)256 * 704 * 2);    // [C][Hpad] max

  const dim3 blk(256);

  // patch embed: (3,384,384) -> x1 (32,384,384), NCHW, written directly to d_out
  k_conv3x3_plain<<<c256(32 * 384 * 384), blk, 0, stream>>>(inp, pe_w, x1o, 3, 32, 384, 384);
  // down 1->2: conv 32->16 then unshuffle -> x2 (64,192,192) NCHW in d_out
  k_conv3x3_down<<<c256(16 * 384 * 384), blk, 0, stream>>>(x1o, d12_w, x2o, 32, 16, 384, 384, 0, 0);
  // down 2->3: conv 64->32 + unshuffle -> act3 NHWC [9216,128]
  k_conv3x3_down<<<c256(32 * 192 * 192), blk, 0, stream>>>(x2o, d23_w, act3, 64, 32, 192, 192, 0, 1);

  // level 3: 3 transformer blocks (dim 128, hidden 340, heads 4)
  for (int j = 0; j < 3; ++j)
    run_block(d_in + 5 + j * 10, act3, 96, 96, 128, 340,
              a16, h16, S1, S2, wq, wpr, wpin, wdw, wpo, stream);

  // x3 output (NCHW)
  k_nhwc_to_nchw<<<c256(128 * 96 * 96), blk, 0, stream>>>(act3, x3o, 128, 96 * 96);
  // down 3->4: conv 128->64 + unshuffle -> act4 NHWC [2304,256]
  k_conv3x3_down<<<c256(64 * 96 * 96), blk, 0, stream>>>(act3, d34_w, act4, 128, 64, 96, 96, 1, 1);

  // level 4: 4 transformer blocks (dim 256, hidden 680, heads 8)
  for (int j = 0; j < 4; ++j)
    run_block(d_in + 35 + j * 10, act4, 48, 48, 256, 680,
              a16, h16, S1, S2, wq, wpr, wpin, wdw, wpo, stream);

  // x4 output (NCHW)
  k_nhwc_to_nchw<<<c256(256 * 48 * 48), blk, 0, stream>>>(act4, x4o, 256, 48 * 48);
}
